// LayerNormMLP_1563368095883
// MI455X (gfx1250) — matrix operand order repaired and checked by reading, not compiled, not run on hardware
//
#include <hip/hip_runtime.h>
#include <hip/hip_bf16.h>

typedef __attribute__((ext_vector_type(16))) __bf16 v16bf;
typedef __attribute__((ext_vector_type(8)))  __bf16 v8bf;
typedef __attribute__((ext_vector_type(8)))  float  v8f;
typedef __attribute__((ext_vector_type(4)))  int    v4i;

#define BM  128
#define BN  64
#define BN2 128
#define BK  64
#define LDA 72  // padded LDS row stride (elements); 144B = 36-bank step, conflict-free

// ---- gfx1250 async global->LDS path (probe-confirmed in round 3) ----
#if defined(__has_builtin)
#if __has_builtin(__builtin_amdgcn_global_load_async_to_lds_b128)
#define HAS_ASYNC_LDS 1
#endif
#if __has_builtin(__builtin_amdgcn_s_wait_asynccnt)
#define HAS_WAIT_ASYNC 1
#endif
#endif

#define GAS __attribute__((address_space(1)))
#define LAS __attribute__((address_space(3)))

__device__ __forceinline__ void cp16(__bf16* l, const __bf16* g) {
#ifdef HAS_ASYNC_LDS
    __builtin_amdgcn_global_load_async_to_lds_b128(
        (GAS v4i*)(uintptr_t)g, (LAS v4i*)(uint32_t)(uintptr_t)l, 0, 0);
#else
    *reinterpret_cast<uint4*>(l) = *reinterpret_cast<const uint4*>(g);
#endif
}

__device__ __forceinline__ void wait_async_lds() {
#ifdef HAS_ASYNC_LDS
#ifdef HAS_WAIT_ASYNC
    __builtin_amdgcn_s_wait_asynccnt(0);
#else
    asm volatile("s_wait_asynccnt 0x0" ::: "memory");
#endif
#endif
}

// gelu_tanh(z) = 0.5 z (1 + tanh(u)) = z * sigmoid(2u), u = 0.79788456(z + 0.044715 z^3)
// branch-free: v_exp + v_rcp, saturates correctly at +/-inf
__device__ __forceinline__ float gelu_fast(float z) {
    float u2 = 1.5957691216057308f * (z + 0.044715f * z * z * z);
    return z * __builtin_amdgcn_rcpf(1.0f + __expf(-u2));
}

__device__ __forceinline__ v16bf cat8(v8bf lo, v8bf hi) {
    return __builtin_shufflevector(lo, hi, 0,1,2,3,4,5,6,7,8,9,10,11,12,13,14,15);
}

// ---------------- transpose + fp32->bf16 cast:  src[R][C] -> dst[C][R] ----------------
__global__ __launch_bounds__(256) void transpose_cast_kernel(
    const float* __restrict__ src, __bf16* __restrict__ dst, int R, int C) {
    __shared__ float tile[32][33];
    int cx = blockIdx.x * 32;
    int ry = blockIdx.y * 32;
    int tx = threadIdx.x;
    int ty = threadIdx.y;
    #pragma unroll
    for (int i = ty; i < 32; i += 8)
        tile[i][tx] = src[(size_t)(ry + i) * C + (cx + tx)];
    __syncthreads();
    #pragma unroll
    for (int i = ty; i < 32; i += 8)
        dst[(size_t)(cx + i) * R + (ry + tx)] = (__bf16)tile[tx][i];
}

// ---------------- fused LayerNorm (fp32) -> bf16 ----------------
__global__ __launch_bounds__(256) void layernorm_kernel(
    const float* __restrict__ x, const float* __restrict__ scale,
    const float* __restrict__ bias, __bf16* __restrict__ out, int H) {
    const int token = blockIdx.x;
    const int tid   = threadIdx.x;
    const float* xr = x + (size_t)token * H;
    const int h0 = tid * 8;

    float4 v0 = reinterpret_cast<const float4*>(xr + h0)[0];
    float4 v1 = reinterpret_cast<const float4*>(xr + h0)[1];
    float xv[8] = {v0.x, v0.y, v0.z, v0.w, v1.x, v1.y, v1.z, v1.w};

    float s = 0.f, ss = 0.f;
    #pragma unroll
    for (int j = 0; j < 8; ++j) { s += xv[j]; ss += xv[j] * xv[j]; }
    #pragma unroll
    for (int off = 16; off > 0; off >>= 1) {
        s  += __shfl_xor(s,  off, 32);
        ss += __shfl_xor(ss, off, 32);
    }
    __shared__ float ps[8], pss[8];
    if ((tid & 31) == 0) { ps[tid >> 5] = s; pss[tid >> 5] = ss; }
    __syncthreads();
    float S = 0.f, SS = 0.f;
    #pragma unroll
    for (int i = 0; i < 8; ++i) { S += ps[i]; SS += pss[i]; }
    const float inv_h = 1.0f / (float)H;
    const float mu    = S * inv_h;
    const float var   = SS * inv_h - mu * mu;
    const float rstd  = rsqrtf(var + 1e-6f);

    v8bf o;
    #pragma unroll
    for (int j = 0; j < 8; ++j)
        o[j] = (__bf16)((xv[j] - mu) * rstd * scale[h0 + j] + bias[h0 + j]);
    *reinterpret_cast<v8bf*>(out + (size_t)token * H + h0) = o;
}

// ---------------- GEMM1 + GEGLU:  act = gelu(A@W1g) * (A@W1l) ----------------
// A: [M][H] bf16, W1T: [2F][H] bf16 N-major, act: [M][F] bf16
__global__ __launch_bounds__(256) void gemm_geglu_kernel(
    const __bf16* __restrict__ A, const __bf16* __restrict__ W1T,
    __bf16* __restrict__ act, int M, int H, int F) {
    constexpr int SA  = BM * LDA;            // 9216 elements
    constexpr int SB  = BN * LDA;            // 4608 elements
    constexpr int SSZ = SA + 2 * SB;
    __shared__ __bf16 smem[2 * SSZ];         // 72 KB

    const int tid      = threadIdx.x;
    const int lane     = tid & 31;
    const int wave     = tid >> 5;
    const int waveM    = wave & 3;           // 4 waves x 32 rows
    const int waveN    = wave >> 2;          // 2 waves x 32 cols
    const int laneHalf = lane & 15;
    const int laneHi   = lane >> 4;
    const int m0 = blockIdx.y * BM;
    const int n0 = blockIdx.x * BN;

    // staging: BK=64 -> 8 x 16B chunks per row; thread handles 4 A rows + 2 rows per B tile
    const int col8 = (tid & 7) * 8;          // 0..56
    const int r32  = tid >> 3;               // 0..31
    const __bf16* gA = A   + (size_t)(m0 + r32) * H + col8;
    const __bf16* gB = W1T + (size_t)(n0 + r32) * H + col8;
    const size_t FH = (size_t)F * H;
    const int lA = r32 * LDA + col8;
    const int lB = r32 * LDA + col8;

    v8f acc0[2][2] = {};
    v8f acc1[2][2] = {};

    auto stage = [&](int p, int k) {
        __bf16* s = smem + p * SSZ;
        #pragma unroll
        for (int j = 0; j < 4; ++j)          // A: rows r32 + 32j
            cp16(s + lA + (32 * LDA) * j, gA + k + (size_t)(32 * H) * j);
        #pragma unroll
        for (int j = 0; j < 2; ++j) {        // B gate + linear: rows r32 + 32j
            cp16(s + SA + lB + (32 * LDA) * j,      gB + k + (size_t)(32 * H) * j);
            cp16(s + SA + SB + lB + (32 * LDA) * j, gB + FH + k + (size_t)(32 * H) * j);
        }
    };

    stage(0, 0);
    wait_async_lds();
    __syncthreads();

    const int KT = H / BK;
    int p = 0;
    for (int kt = 0; kt < KT; ++kt) {
        if (kt + 1 < KT) stage(p ^ 1, (kt + 1) * BK);

        const __bf16* s = smem + p * SSZ;
        #pragma unroll
        for (int h = 0; h < 2; ++h) {        // two K=32 halves of the K=64 stage
            const int kh = h * 32;
            v16bf aF[2];
            #pragma unroll
            for (int mi = 0; mi < 2; ++mi) {
                const __bf16* base = s + (waveM * 32 + mi * 16 + laneHalf) * LDA + kh;
                aF[mi] = cat8(*reinterpret_cast<const v8bf*>(base + laneHi * 16),
                              *reinterpret_cast<const v8bf*>(base + 8 + laneHi * 16));
            }
            #pragma unroll
            for (int ni = 0; ni < 2; ++ni) {
                const int coff = (waveN * 32 + ni * 16 + laneHalf) * LDA + kh + laneHi * 16;
                const __bf16* b0 = s + SA + coff;
                const __bf16* b1 = s + SA + SB + coff;
                v16bf bFg = cat8(*reinterpret_cast<const v8bf*>(b0),
                                 *reinterpret_cast<const v8bf*>(b0 + 8));
                v16bf bFl = cat8(*reinterpret_cast<const v8bf*>(b1),
                                 *reinterpret_cast<const v8bf*>(b1 + 8));
                #pragma unroll
                for (int mi = 0; mi < 2; ++mi) {
                    acc0[mi][ni] = __builtin_amdgcn_wmma_f32_16x16x32_bf16(
                        false, aF[mi], false, bFg, (short)0, acc0[mi][ni], false, false);
                    acc1[mi][ni] = __builtin_amdgcn_wmma_f32_16x16x32_bf16(
                        false, aF[mi], false, bFl, (short)0, acc1[mi][ni], false, false);
                }
            }
        }
        wait_async_lds();
        __syncthreads();
        p ^= 1;
    }

    #pragma unroll
    for (int mi = 0; mi < 2; ++mi) {
        #pragma unroll
        for (int ni = 0; ni < 2; ++ni) {
            const int rowBase = m0 + waveM * 32 + mi * 16 + laneHi * 8;
            const int col     = n0 + waveN * 32 + ni * 16 + laneHalf;
            #pragma unroll
            for (int r = 0; r < 8; ++r) {
                float g = gelu_fast(acc0[mi][ni][r]) * acc1[mi][ni][r];
                act[(size_t)(rowBase + r) * F + col] = (__bf16)g;
            }
        }
    }
}

// ---------------- GEMM2:  out = act @ kernel_2  (fp32 out), 128x128 tile ----------------
// A: [M][F] bf16, W2T: [H][F] bf16 N-major, out: [M][H] fp32
__global__ __launch_bounds__(256) void gemm2_kernel(
    const __bf16* __restrict__ A, const __bf16* __restrict__ W2T,
    float* __restrict__ out, int M, int F, int H) {
    constexpr int SA  = BM  * LDA;
    constexpr int SB  = BN2 * LDA;
    constexpr int SSZ = SA + SB;
    __shared__ __bf16 smem[2 * SSZ];         // 72 KB

    const int tid      = threadIdx.x;
    const int lane     = tid & 31;
    const int wave     = tid >> 5;
    const int waveM    = wave & 1;           // 2 waves x 64 rows
    const int waveN    = wave >> 1;          // 4 waves x 32 cols
    const int laneHalf = lane & 15;
    const int laneHi   = lane >> 4;
    const int m0 = blockIdx.y * BM;
    const int n0 = blockIdx.x * BN2;

    const int col8 = (tid & 7) * 8;
    const int r32  = tid >> 3;               // 0..31
    const __bf16* gA = A   + (size_t)(m0 + r32) * F + col8;
    const __bf16* gB = W2T + (size_t)(n0 + r32) * F + col8;
    const int lA = r32 * LDA + col8;

    v8f acc[4][2] = {};

    auto stage = [&](int p, int k) {
        __bf16* s = smem + p * SSZ;
        #pragma unroll
        for (int j = 0; j < 4; ++j) {        // A rows r32+32j, B rows r32+32j
            cp16(s + lA + (32 * LDA) * j,      gA + k + (size_t)(32 * F) * j);
            cp16(s + SA + lA + (32 * LDA) * j, gB + k + (size_t)(32 * F) * j);
        }
    };

    stage(0, 0);
    wait_async_lds();
    __syncthreads();

    const int KT = F / BK;
    int p = 0;
    for (int kt = 0; kt < KT; ++kt) {
        if (kt + 1 < KT) stage(p ^ 1, (kt + 1) * BK);

        const __bf16* s = smem + p * SSZ;
        #pragma unroll
        for (int h = 0; h < 2; ++h) {
            const int kh = h * 32;
            v16bf aF[4];
            #pragma unroll
            for (int mi = 0; mi < 4; ++mi) {
                const __bf16* base = s + (waveM * 64 + mi * 16 + laneHalf) * LDA + kh;
                aF[mi] = cat8(*reinterpret_cast<const v8bf*>(base + laneHi * 16),
                              *reinterpret_cast<const v8bf*>(base + 8 + laneHi * 16));
            }
            #pragma unroll
            for (int ni = 0; ni < 2; ++ni) {
                const __bf16* b =
                    s + SA + (waveN * 32 + ni * 16 + laneHalf) * LDA + kh + laneHi * 16;
                v16bf bF = cat8(*reinterpret_cast<const v8bf*>(b),
                                *reinterpret_cast<const v8bf*>(b + 8));
                #pragma unroll
                for (int mi = 0; mi < 4; ++mi)
                    acc[mi][ni] = __builtin_amdgcn_wmma_f32_16x16x32_bf16(
                        false, aF[mi], false, bF, (short)0, acc[mi][ni], false, false);
            }
        }
        wait_async_lds();
        __syncthreads();
        p ^= 1;
    }

    #pragma unroll
    for (int mi = 0; mi < 4; ++mi) {
        #pragma unroll
        for (int ni = 0; ni < 2; ++ni) {
            const int rowBase = m0 + waveM * 64 + mi * 16 + laneHi * 8;
            const int col     = n0 + waveN * 32 + ni * 16 + laneHalf;
            #pragma unroll
            for (int r = 0; r < 8; ++r)
                out[(size_t)(rowBase + r) * H + col] = acc[mi][ni][r];
        }
    }
}

extern "C" void kernel_launch(void* const* d_in, const int* in_sizes, int n_in,
                              void* d_out, int out_size, void* d_ws, size_t ws_size,
                              hipStream_t stream) {
    const float* x     = (const float*)d_in[0];   // [S,B,H]
    const float* scale = (const float*)d_in[1];   // [H]
    const float* lnb   = (const float*)d_in[2];   // [H]
    const float* w1    = (const float*)d_in[3];   // [H][2F]
    const float* w2    = (const float*)d_in[4];   // [F][H]
    float* out = (float*)d_out;

    const int S = 2048, B = 4, H = 2048, F = 8192;
    const int M = S * B;

    char* ws = (char*)d_ws;
    __bf16* lnB = (__bf16*)(ws);                       // 32MB
    __bf16* W1T = (__bf16*)(ws + ((size_t)32  << 20)); // 64MB
    __bf16* W2T = (__bf16*)(ws + ((size_t)96  << 20)); // 32MB
    __bf16* act = (__bf16*)(ws + ((size_t)128 << 20)); // 128MB

    transpose_cast_kernel<<<dim3((2 * F) / 32, H / 32), dim3(32, 8), 0, stream>>>(
        w1, W1T, H, 2 * F);
    transpose_cast_kernel<<<dim3(H / 32, F / 32), dim3(32, 8), 0, stream>>>(
        w2, W2T, F, H);
    layernorm_kernel<<<M, 256, 0, stream>>>(x, scale, lnb, lnB, H);
    gemm_geglu_kernel<<<dim3(F / BN, M / BM), 256, 0, stream>>>(lnB, W1T, act, M, H, F);
    gemm2_kernel<<<dim3(H / BN2, M / BM), 256, 0, stream>>>(act, W2T, out, M, F, H);
}
